// CustomMultiHeadAttention_70858370449595
// MI455X (gfx1250) — matrix operand order repaired and checked by reading, not compiled, not run on hardware
//
#include <hip/hip_runtime.h>
#include <hip/hip_bf16.h>

typedef __attribute__((ext_vector_type(16))) _Float16 v16h;
typedef __attribute__((ext_vector_type(8)))  _Float16 v8h;
typedef __attribute__((ext_vector_type(8)))  float    v8f;
typedef __attribute__((ext_vector_type(4)))  float    v4f;

constexpr int B_ = 2;
constexpr int S_ = 2048;
constexpr int E_ = 1024;
constexpr int H_ = 16;
constexpr int D_ = 64;
constexpr int MTOT = B_ * S_;          // 4096 rows in projection GEMMs
constexpr int LDS_STRIDE = 2052;       // 2048 + 4 pad: breaks 64-bank alignment

// ---------------------------------------------------------------------------
// WMMA fragment loaders (wave32, 16x16x32 f16 shape).
// A (16x32, MxK): lane&15 = M row; lane>>4 selects K group of 16.
// B (32x16, KxN): lane&15 = N col; lane>>4 selects K group of 16.
// D (16x16):      lane&15 = N; M = (lane>>4)*8 + i for v8f element i.
// ---------------------------------------------------------------------------

__device__ inline v16h load_a_f32(const float* base, int ld, int k0, int lane) {
  const int m  = lane & 15;
  const int ko = (lane >> 4) * 16;
  const float* p = base + (size_t)m * ld + k0 + ko;
  v4f x0 = *(const v4f*)(p + 0);
  v4f x1 = *(const v4f*)(p + 4);
  v4f x2 = *(const v4f*)(p + 8);
  v4f x3 = *(const v4f*)(p + 12);
  v16h a;
#pragma unroll
  for (int i = 0; i < 4; ++i) {
    a[i + 0]  = (_Float16)x0[i];
    a[i + 4]  = (_Float16)x1[i];
    a[i + 8]  = (_Float16)x2[i];
    a[i + 12] = (_Float16)x3[i];
  }
  return a;
}

__device__ inline v16h load_b_f32(const float* __restrict__ base, int ld,
                                  int k0, int lane) {
  const int n  = lane & 15;
  const int ko = (lane >> 4) * 16;
  const float* p = base + (size_t)n * ld + k0 + ko;
  v16h b;
#pragma unroll
  for (int c = 0; c < 4; ++c) {
    v4f x = *(const v4f*)(p + 4 * c);
#pragma unroll
    for (int i = 0; i < 4; ++i) b[4 * c + i] = (_Float16)x[i];
  }
  return b;
}

__device__ inline v16h load_a_f16(const _Float16* __restrict__ base, int ld,
                                  int k0, int lane) {
  const int m  = lane & 15;
  const int ko = (lane >> 4) * 16;
  const _Float16* p = base + (size_t)m * ld + k0 + ko;
  v8h c0 = *(const v8h*)(p);
  v8h c1 = *(const v8h*)(p + 8);
  v16h a;
#pragma unroll
  for (int i = 0; i < 8; ++i) { a[i] = c0[i]; a[8 + i] = c1[i]; }
  return a;
}

__device__ inline v16h load_b_f16(const _Float16* __restrict__ base, int ld,
                                  int k0, int lane) {
  const int n  = lane & 15;
  const int ko = (lane >> 4) * 16;
  const _Float16* p = base + (size_t)n * ld + k0 + ko;
  return *(const v16h*)(p);
}

__device__ inline v8f wmma_f16(const v16h& a, const v16h& b, const v8f& c) {
  return __builtin_amdgcn_wmma_f32_16x16x32_f16(
      /*neg_a=*/false, a, /*neg_b=*/false, b,
      /*c_mod=*/(short)0, c, /*reuse_a=*/false, /*reuse_b=*/false);
}

// ---------------------------------------------------------------------------
// Kernel 1: fused QKV projection.  X(4096x1024) @ W^T(1024x1024) + bias.
// z = 0/1/2 -> Q/K/V.  Q,K stored f16 as (b,h,s,d); V stored f16 as (b,h,d,s).
// ---------------------------------------------------------------------------
__global__ __launch_bounds__(256) void mha_proj_kernel(
    const float* __restrict__ xq, const float* __restrict__ xk,
    const float* __restrict__ xv,
    const float* __restrict__ Wq, const float* __restrict__ Wk,
    const float* __restrict__ Wv,
    const float* __restrict__ bq, const float* __restrict__ bk,
    const float* __restrict__ bv,
    _Float16* __restrict__ q16, _Float16* __restrict__ k16,
    _Float16* __restrict__ vt16) {
  const int lane  = threadIdx.x & 31;
  const int wid   = threadIdx.x >> 5;
  const int wgid  = blockIdx.x * 8 + wid;   // 16384 waves: 256 mTiles x 64 nTiles
  const int mTile = wgid >> 6;
  const int nTile = wgid & 63;
  const int m0 = mTile * 16, n0 = nTile * 16;
  const int z = blockIdx.z;

  const float* X = (z == 0) ? xq : (z == 1) ? xk : xv;
  const float* W = (z == 0) ? Wq : (z == 1) ? Wk : Wv;
  const float* bb = (z == 0) ? bq : (z == 1) ? bk : bv;

  const float* aBase = X + (size_t)m0 * E_;
  const float* wBase = W + (size_t)n0 * E_;

  v8f c = {};
  for (int k0 = 0; k0 < E_; k0 += 32) {
    __builtin_prefetch((const void*)(aBase + (size_t)(lane & 15) * E_ + k0 + 128), 0, 1);
    v16h a = load_a_f32(aBase, E_, k0, lane);
    v16h b = load_b_f32(wBase, E_, k0, lane);
    c = wmma_f16(a, b, c);
  }

  const int n  = lane & 15;
  const int mb = (lane >> 4) * 8;
  const int gn = n0 + n;
  const int h  = gn >> 6;       // head
  const int d  = gn & 63;       // dim within head
  const int bIdx = m0 >> 11;    // batch (16-row tile never crosses S=2048)
  const float bval = bb[gn];

#pragma unroll
  for (int i = 0; i < 8; ++i) {
    const int m = m0 + mb + i;
    const int s = m & (S_ - 1);
    const _Float16 val = (_Float16)(c[i] + bval);
    if (z == 2) {
      vt16[(((size_t)bIdx * H_ + h) * D_ + d) * S_ + s] = val;   // (b,h,d,s)
    } else {
      _Float16* dst = (z == 0) ? q16 : k16;
      dst[(((size_t)bIdx * H_ + h) * S_ + s) * D_ + d] = val;    // (b,h,s,d)
    }
  }
}

// ---------------------------------------------------------------------------
// Kernel 2: FUSED scores + softmax + PV.
// One 256-thread block (8 waves) owns a 16-row s_q stripe of one (b,h):
//   phase 1: all 16x2048 scores via WMMA into a 128KB LDS tile (needs CDNA5's
//            320KB-per-WGP LDS),
//   phase 2: row max / exp / sum reductions in LDS; normalized f32 attn
//            written to HBM exactly once (mandatory output, coalesced),
//   phase 3: ctx = P @ V computed straight from the LDS tile (A-fragments are
//            ds_load'ed f32 exp values converted to f16 in registers; 1/rowSum
//            folded into the epilogue).  8 waves = 4 d-tiles x 2 k-halves,
//            split-K partials reduced through LDS.
// This removes both the separate softmax pass and the 537MB attn re-read.
// ---------------------------------------------------------------------------
__global__ __launch_bounds__(256) void mha_scores_softmax_pv_kernel(
    const _Float16* __restrict__ q16, const _Float16* __restrict__ k16,
    const _Float16* __restrict__ vt16, float* __restrict__ attn,
    _Float16* __restrict__ ctx16) {
  extern __shared__ float smem[];
  float* tile    = smem;                    // [16][LDS_STRIDE] scores -> exp
  float* red     = smem + 16 * LDS_STRIDE;  // [256] partial reductions
  float* rowStat = red + 256;               // [16] row max, then row 1/sum
  float* psum    = rowStat + 16;            // [8][256] PV split-K partials

  const int t    = threadIdx.x;
  const int lane = t & 31;
  const int wid  = t >> 5;
  const int z  = blockIdx.z;                // b*H + h
  const int m0 = blockIdx.x * 16;           // s_q tile

  const _Float16* qB    = q16 + ((size_t)z * S_ + m0) * D_;
  const _Float16* kBase = k16 + (size_t)z * S_ * D_;

  // A fragments (the 16 q rows, K = D = 64 -> two k-steps): load once,
  // reuse across all 16 n-tiles this wave computes.
  const v16h a0 = load_a_f16(qB, D_, 0, lane);
  const v16h a1 = load_a_f16(qB, D_, 32, lane);

  const int n  = lane & 15;
  const int mb = (lane >> 4) * 8;

  // --- phase 1: scores.  Each wave covers 256 consecutive s_k columns. ---
  for (int j = 0; j < 16; ++j) {
    const int n0 = wid * 256 + j * 16;
    const _Float16* kB = kBase + (size_t)n0 * D_;
    __builtin_prefetch((const void*)(kB + 16 * D_), 0, 1);
    v8f c = {};
    c = wmma_f16(a0, load_b_f16(kB, D_, 0, lane), c);
    c = wmma_f16(a1, load_b_f16(kB, D_, 32, lane), c);
#pragma unroll
    for (int i = 0; i < 8; ++i)
      tile[(mb + i) * LDS_STRIDE + n0 + n] = c[i] * 0.125f;  // D^-0.5 = 1/8
  }
  __syncthreads();

  // --- phase 2: row softmax stats (16 threads per row). ---
  const int r   = t >> 4;   // row 0..15
  const int c16 = t & 15;

  float mx = -3.0e38f;
  for (int i = c16; i < S_; i += 16) mx = fmaxf(mx, tile[r * LDS_STRIDE + i]);
  red[t] = mx;
  __syncthreads();
  if (c16 == 0) {
    float m2 = red[t];
#pragma unroll
    for (int i = 1; i < 16; ++i) m2 = fmaxf(m2, red[t + i]);
    rowStat[r] = m2;
  }
  __syncthreads();

  const float rmx = rowStat[r];
  float sum = 0.0f;
  for (int i = c16; i < S_; i += 16) {
    float e = __expf(tile[r * LDS_STRIDE + i] - rmx);
    tile[r * LDS_STRIDE + i] = e;
    sum += e;
  }
  red[t] = sum;
  __syncthreads();
  if (c16 == 0) {
    float s2 = 0.0f;
#pragma unroll
    for (int i = 0; i < 16; ++i) s2 += red[t + i];
    rowStat[r] = 1.0f / s2;
  }
  __syncthreads();

  // Single coalesced write of the normalized attention stripe (tile keeps
  // un-normalized exp; scale applied on the way out).
  float* outB = attn + ((size_t)z * S_ + m0) * S_;
  for (int idx = t; idx < 16 * S_; idx += 256) {
    const int rr = idx >> 11;        // / 2048
    const int cc = idx & (S_ - 1);
    outB[(size_t)rr * S_ + cc] = tile[rr * LDS_STRIDE + cc] * rowStat[rr];
  }

  // --- phase 3: ctx = P @ V from the LDS tile. ---
  const int dTile = wid & 3;        // which 16-wide slice of D
  const int kHalf = wid >> 2;       // split-K half of S
  const _Float16* vB = vt16 + ((size_t)z * D_ + dTile * 16) * S_;

  v8f cc = {};
  for (int k0 = kHalf * (S_ / 2); k0 < kHalf * (S_ / 2) + S_ / 2; k0 += 32) {
    v16h a = load_a_f32(tile, LDS_STRIDE, k0, lane);   // ds_load + cvt
    v16h b = load_b_f16(vB, S_, k0, lane);
    cc = wmma_f16(a, b, cc);
  }
#pragma unroll
  for (int i = 0; i < 8; ++i)
    psum[wid * 256 + (mb + i) * 16 + n] = cc[i];
  __syncthreads();

  // Reduce the two k-halves, fold in 1/rowSum, store f16 ctx (b, s, h*D+d).
  const int bIdx = z / H_, h = z % H_;
  for (int e = t; e < 4 * 256; e += 256) {
    const int dt  = e >> 8;          // d-tile 0..3
    const int ee  = e & 255;
    const int row = ee >> 4;
    const int col = ee & 15;
    const float vsum = psum[dt * 256 + ee] + psum[(dt + 4) * 256 + ee];
    ctx16[((size_t)bIdx * S_ + m0 + row) * E_ + h * D_ + dt * 16 + col] =
        (_Float16)(vsum * rowStat[row]);
  }
}

// ---------------------------------------------------------------------------
// Kernel 3: out = ctx @ Wo^T + bo, f32 into d_out[0 : B*S*E).
// ---------------------------------------------------------------------------
__global__ __launch_bounds__(256) void mha_out_kernel(
    const _Float16* __restrict__ ctx16, const float* __restrict__ Wo,
    const float* __restrict__ bo, float* __restrict__ out) {
  const int lane  = threadIdx.x & 31;
  const int wid   = threadIdx.x >> 5;
  const int wgid  = blockIdx.x * 8 + wid;
  const int mTile = wgid >> 6;
  const int nTile = wgid & 63;
  const int m0 = mTile * 16, n0 = nTile * 16;

  const _Float16* aB = ctx16 + (size_t)m0 * E_;
  const float*    wB = Wo    + (size_t)n0 * E_;

  v8f c = {};
  for (int k0 = 0; k0 < E_; k0 += 32) {
    v16h a = load_a_f16(aB, E_, k0, lane);
    v16h b = load_b_f32(wB, E_, k0, lane);
    c = wmma_f16(a, b, c);
  }

  const int n  = lane & 15;
  const int mb = (lane >> 4) * 8;
  const float bval = bo[n0 + n];
#pragma unroll
  for (int i = 0; i < 8; ++i)
    out[(size_t)(m0 + mb + i) * E_ + n0 + n] = c[i] + bval;
}

// ---------------------------------------------------------------------------
extern "C" void kernel_launch(void* const* d_in, const int* in_sizes, int n_in,
                              void* d_out, int out_size, void* d_ws, size_t ws_size,
                              hipStream_t stream) {
  const float* q  = (const float*)d_in[0];
  const float* k  = (const float*)d_in[1];
  const float* v  = (const float*)d_in[2];
  const float* Wq = (const float*)d_in[3];
  const float* bq = (const float*)d_in[4];
  const float* Wk = (const float*)d_in[5];
  const float* bk = (const float*)d_in[6];
  const float* Wv = (const float*)d_in[7];
  const float* bv = (const float*)d_in[8];
  const float* Wo = (const float*)d_in[9];
  const float* bo = (const float*)d_in[10];

  float* out  = (float*)d_out;
  float* attn = out + (size_t)B_ * S_ * E_;     // second tuple output

  _Float16* ws    = (_Float16*)d_ws;            // 32 MB of f16 scratch
  _Float16* q16   = ws;
  _Float16* k16   = q16  + (size_t)B_ * H_ * S_ * D_;
  _Float16* vt16  = k16  + (size_t)B_ * H_ * S_ * D_;
  _Float16* ctx16 = vt16 + (size_t)B_ * H_ * S_ * D_;

  // 1) QKV projections: 16384 16x16 tiles per matrix, 8 waves/block.
  dim3 gProj(MTOT / 16 * (E_ / 16) / 8, 1, 3);  // (2048,1,3)
  mha_proj_kernel<<<gProj, 256, 0, stream>>>(q, k, v, Wq, Wk, Wv, bq, bk, bv,
                                             q16, k16, vt16);

  // 2) fused scores + softmax + PV: one block per (b,h, 16-row q stripe).
  const size_t ldsBytes =
      (size_t)(16 * LDS_STRIDE + 256 + 16 + 8 * 256) * sizeof(float);
  dim3 gSc(S_ / 16, 1, B_ * H_);                // (128,1,32)
  mha_scores_softmax_pv_kernel<<<gSc, 256, ldsBytes, stream>>>(q16, k16, vt16,
                                                               attn, ctx16);

  // 3) out = ctx @ Wo^T + bo
  dim3 gOut(MTOT / 16 * (E_ / 16) / 8, 1, 1);   // (2048,1,1)
  mha_out_kernel<<<gOut, 256, 0, stream>>>(ctx16, Wo, bo, out);
}
